// ConvBN2d__9311489098304
// MI455X (gfx1250) — matrix operand order repaired and checked by reading, not compiled, not run on hardware
//
#include <hip/hip_runtime.h>
#include <hip/hip_bf16.h>

// ---------------- problem constants ----------------
#define N_    8
#define T_    20
#define CIN_  32
#define COUT_ 64
#define H_    64
#define W_    64
#define OH_   32
#define OW_   32
#define KDIM  288          // CIN * 3 * 3, ordered k = tap*32 + cin
#define EPS_  1e-5f

// ---------------- block tiling ----------------
#define NTHREADS 512       // 16 wave32 per block
#define TR    8            // pre-pool rows per block tile
#define TC    32           // pre-pool cols per block tile
#define XROWS 10           // TR + halo
#define XCOLS 34           // TC + halo
#define NPOS  256          // TR*TC positions (GEMM N per block)
#define X_ELEMS (XROWS*XCOLS*CIN_)   // 10880

// ---------------- dynamic LDS layout (bytes) ----------------
#define W_OFF   0
#define W_BYTES (COUT_*KDIM*2)        // 36864 bf16 weights
#define B_OFF   (W_OFF + W_BYTES)
#define B_BYTES (COUT_*4)             // 256  fp32 bias
#define X_OFF   (B_OFF + B_BYTES)
#define X_BYTES (X_ELEMS*2)           // 21760 bf16 input tile (x2: double buffered)
#define C_OFF   (X_OFF + 2*X_BYTES)
#define C_BYTES (COUT_*NPOS*4)        // 65536 fp32 conv staging for pooling
#define SMEM_BYTES (C_OFF + C_BYTES)  // 146176 (2 blocks fit in 320KB WGP LDS)

typedef __attribute__((ext_vector_type(16))) __bf16 v16bf;
typedef __attribute__((ext_vector_type(8)))  float  v8f;

// ================= BN folding / weight repack =================
__global__ void prep_weights(const float* __restrict__ w, const float* __restrict__ b,
                             const float* __restrict__ gamma, const float* __restrict__ beta,
                             const float* __restrict__ rm, const float* __restrict__ rv,
                             __bf16* __restrict__ wS, __bf16* __restrict__ wA,
                             float* __restrict__ bS, float* __restrict__ bA)
{
    int idx = blockIdx.x * blockDim.x + threadIdx.x;
    if (idx < COUT_ * KDIM) {
        int cout = idx / KDIM;
        int k    = idx - cout * KDIM;
        int tap  = k >> 5;           // k / 32
        int cin  = k & 31;
        int kh = tap / 3, kw = tap % 3;
        float wv = w[((cout*CIN_ + cin)*3 + kh)*3 + kw];
        float rS = gamma[cout] * rsqrtf(rv[cout]);          // spiking path: no eps (matches ref)
        float rA = gamma[cout] * rsqrtf(rv[cout] + EPS_);   // ANN path: eps
        int didx = (cout*9 + tap)*CIN_ + cin;               // [cout][tap][cin]
        wS[didx] = (__bf16)(wv * rS);
        wA[didx] = (__bf16)(wv * rA);
    }
    if (idx < COUT_) {
        float rS = gamma[idx] * rsqrtf(rv[idx]);
        float rA = gamma[idx] * rsqrtf(rv[idx] + EPS_);
        bS[idx] = (b[idx] - rm[idx]) * rS + beta[idx];
        bA[idx] = (b[idx] - rm[idx]) * rA + beta[idx];
    }
}

// ---- stage one input tile (zero-padded halo) as bf16 into LDS [row][col][cin] ----
__device__ __forceinline__ void stage_tile(const float* __restrict__ src,
                                           __bf16* __restrict__ xb,
                                           int tid, int row0, int col0)
{
    #pragma unroll 2
    for (int idx = tid; idx < X_ELEMS; idx += NTHREADS) {
        int cin = idx / (XROWS*XCOLS);
        int rem = idx - cin * (XROWS*XCOLS);
        int r   = rem / XCOLS;
        int c   = rem - r * XCOLS;
        int hh  = row0 + r - 1;
        int ww  = col0 + c - 1;
        float v = 0.f;
        if (hh >= 0 && hh < H_ && ww >= 0 && ww < W_)
            v = src[(cin*H_ + hh)*W_ + ww];
        xb[(r*XCOLS + c)*CIN_ + cin] = (__bf16)v;
    }
}

// ================= fused conv + pool (+ spike scan | relu) =================
// grid = (16 tiles, N_), block = 512 threads (16 wave32).
// Per timestep GEMM: M=64 (cout) x N=256 (positions) x K=288 (9 taps * 32 cin)
// via v_wmma_f32_16x16x32_bf16. Wave w: mt = w&3 (cout 16-tile), quad = w>>2
// (64 positions = 4 N-tiles = 2 pre-pool rows). Input tile is double buffered
// in LDS so global->LDS staging of t+1 overlaps the WMMA GEMM of t.
// All fragment addresses are constant offsets from per-wave base pointers,
// folding into ds_load_b128 immediate offsets.
template<bool SPIKE>
__global__ __launch_bounds__(NTHREADS) void conv_pool_kernel(
    const float* __restrict__ x,        // SPIKE: (N,T,CIN,H,W)  else (N,CIN,H,W)
    const __bf16* __restrict__ wq,      // bf16 [cout][tap][cin]
    const float* __restrict__ bq,       // fp32 [cout]
    float* __restrict__ out_main,       // SPIKE: spike_out (N,T,COUT,OH,OW) else ann (N,COUT,OH,OW)
    float* __restrict__ out_count)      // SPIKE: (N,COUT,OH,OW)
{
    extern __shared__ char smem[];
    __bf16* w_lds  = (__bf16*)(smem + W_OFF);
    float*  b_lds  = (float* )(smem + B_OFF);
    __bf16* x_lds0 = (__bf16*)(smem + X_OFF);
    __bf16* x_lds1 = (__bf16*)(smem + X_OFF + X_BYTES);
    float*  c_lds  = (float* )(smem + C_OFF);

    const int tid  = threadIdx.x;
    const int lane = tid & 31;
    const int wv   = tid >> 5;      // 0..15
    const int mt   = wv & 3;        // cout tile (16 couts)
    const int quad = wv >> 2;       // 64 positions = 2 pre-pool rows
    const int n16  = lane & 15;
    const int hi   = lane >> 4;

    const int tileid = blockIdx.x;  // 0..15
    const int nimg   = blockIdx.y;  // 0..7
    const int trow = tileid >> 1;   // 0..7
    const int tcol = tileid & 1;    // 0..1
    const int row0 = trow * TR;
    const int col0 = tcol * TC;
    const int oh0  = row0 >> 1;
    const int ow0  = col0 >> 1;

    const size_t isz = (size_t)CIN_ * H_ * W_;
    const float* src0 = SPIKE ? (x + (size_t)nimg * T_ * isz) : (x + (size_t)nimg * isz);

    // ---- stage folded weights + bias + first input tile, one barrier ----
    {
        const uint4* srcw = reinterpret_cast<const uint4*>(wq);
        uint4*       dstw = reinterpret_cast<uint4*>(w_lds);
        const int n16b = W_BYTES / 16;              // 2304
        #pragma unroll
        for (int i = 0; i < (n16b + NTHREADS - 1) / NTHREADS; ++i) {
            int j = tid + i * NTHREADS;
            if (j < n16b) dstw[j] = srcw[j];
        }
        if (tid < COUT_) b_lds[tid] = bq[tid];
    }
    stage_tile(src0, x_lds0, tid, row0, col0);
    __syncthreads();

    // ---- per-wave constant base pointers (everything else is imm offsets) ----
    // A frag: 16x32 bf16; lane m=n16, hi selects K {+8}; chunks {0..7},{16..23}
    const __bf16* wbase  = w_lds + (mt*16 + n16)*KDIM + hi*8;
    // B frag: 32x16 bf16; col=n16; hi selects K {+8}; chunks {0..7},{16..23}
    const __bf16* xbase0 = x_lds0 + (quad*2*XCOLS + n16)*CIN_ + hi*8;
    const __bf16* xbase1 = x_lds1 + (quad*2*XCOLS + n16)*CIN_ + hi*8;
    // C writeback: c_lds[cout][p], cout = mt*16 + e + 8*hi, p = quad*64 + nt*16 + n16
    float* cwb = c_lds + (mt*16 + 8*hi)*NPOS + quad*64 + n16;
    // bias per C row, loop-invariant across t
    float biasr[8];
    #pragma unroll
    for (int e = 0; e < 8; ++e) biasr[e] = b_lds[mt*16 + e + 8*hi];

    // pooling / output geometry (j advances cout by 8)
    const int cout0 = tid >> 6;          // 0..7
    const int oh_t  = (tid >> 4) & 3;    // 0..3
    const int ow_t  = tid & 15;          // 0..15
    const float* cpool = c_lds + cout0*NPOS + (2*oh_t)*TC + 2*ow_t;
    const int ohg = oh0 + oh_t, owg = ow0 + ow_t;

    // per-thread spiking state: 8 output elements each (64*4*16 = 4096 per block)
    float pot[8], mask[8], cnt[8];
    if (SPIKE) {
        #pragma unroll
        for (int j = 0; j < 8; ++j) { pot[j] = 0.f; mask[j] = 0.f; cnt[j] = 0.f; }
    }

    const int tsteps = SPIKE ? T_ : 1;
    for (int t = 0; t < tsteps; ++t) {
        // ---- stage next tile into the other buffer; overlaps this GEMM ----
        if (t + 1 < tsteps)
            stage_tile(src0 + (size_t)(t + 1) * isz, (t & 1) ? x_lds0 : x_lds1,
                       tid, row0, col0);
        if (SPIKE && (t + 2 < T_))  // global_prefetch_b8 two steps ahead (L2 warm)
            __builtin_prefetch(src0 + (size_t)(t + 2) * isz + tid * 16, 0, 1);

        const __bf16* xb = (t & 1) ? xbase1 : xbase0;

        // ---- implicit GEMM ----
        v8f acc[4];
        {
            v8f z = {};
            #pragma unroll
            for (int nt = 0; nt < 4; ++nt) acc[nt] = z;
        }

        #pragma unroll
        for (int tap = 0; tap < 9; ++tap) {
            const int kh = tap / 3, kw = tap % 3;
            union { v16bf v; uint4 q[2]; } afr;
            afr.q[0] = *reinterpret_cast<const uint4*>(wbase + tap*32);
            afr.q[1] = *reinterpret_cast<const uint4*>(wbase + tap*32 + 16);

            #pragma unroll
            for (int nt = 0; nt < 4; ++nt) {
                const int off = (((nt >> 1) + kh)*XCOLS + (nt & 1)*16 + kw)*CIN_;
                union { v16bf v; uint4 q[2]; } bfr;
                bfr.q[0] = *reinterpret_cast<const uint4*>(xb + off);
                bfr.q[1] = *reinterpret_cast<const uint4*>(xb + off + 16);
                acc[nt] = __builtin_amdgcn_wmma_f32_16x16x32_bf16(
                    false, afr.v, false, bfr.v, (short)0, acc[nt], false, false);
            }
        }

        // ---- write C (+bias) to LDS for cross-lane 2x2 pooling ----
        #pragma unroll
        for (int nt = 0; nt < 4; ++nt) {
            #pragma unroll
            for (int e = 0; e < 8; ++e)
                cwb[e*NPOS + nt*16] = acc[nt][e] + biasr[e];
        }
        __syncthreads();

        // ---- 2x2 maxpool + (spike scan | relu) ----
        #pragma unroll
        for (int j = 0; j < 8; ++j) {
            const float2 p0 = *reinterpret_cast<const float2*>(cpool + j*8*NPOS);
            const float2 p1 = *reinterpret_cast<const float2*>(cpool + j*8*NPOS + TC);
            float m = fmaxf(fmaxf(p0.x, p0.y), fmaxf(p1.x, p1.y));
            const int cout = j*8 + cout0;
            if (SPIKE) {
                pot[j] += m;
                float s = (pot[j] >= 1.0f) ? (1.0f - mask[j]) : 0.0f;  // (pot>=1)*(1-mask)
                pot[j] -= s;
                mask[j] = (mask[j] + s > 0.0f) ? 1.0f : 0.0f;
                cnt[j] += s;
                out_main[((((size_t)nimg*T_ + t)*COUT_ + cout)*OH_ + ohg)*OW_ + owg] = s;
            } else {
                out_main[(((size_t)nimg*COUT_ + cout)*OH_ + ohg)*OW_ + owg] = fmaxf(m, 0.0f);
            }
        }
        __syncthreads();   // c_lds reads (this t) vs c_lds writes (next t)
    }

    if (SPIKE) {
        #pragma unroll
        for (int j = 0; j < 8; ++j) {
            const int cout = j*8 + cout0;
            out_count[(((size_t)nimg*COUT_ + cout)*OH_ + ohg)*OW_ + owg] = cnt[j];
        }
    }
}

// ================= launch =================
extern "C" void kernel_launch(void* const* d_in, const int* in_sizes, int n_in,
                              void* d_out, int out_size, void* d_ws, size_t ws_size,
                              hipStream_t stream)
{
    (void)in_sizes; (void)n_in; (void)out_size; (void)ws_size;

    const float* x_st  = (const float*)d_in[0];
    const float* x_sc  = (const float*)d_in[1];
    const float* wgt   = (const float*)d_in[2];
    const float* bias  = (const float*)d_in[3];
    const float* gamma = (const float*)d_in[4];
    const float* beta  = (const float*)d_in[5];
    const float* rmean = (const float*)d_in[6];
    const float* rvar  = (const float*)d_in[7];

    // workspace: folded bf16 weights (x2) + fp32 biases (x2)
    char* ws = (char*)d_ws;
    __bf16* wS = (__bf16*)(ws + 0);
    __bf16* wA = (__bf16*)(ws + W_BYTES);
    float*  bS = (float*)(ws + 2*W_BYTES);
    float*  bA = (float*)(ws + 2*W_BYTES + 256);

    float* out        = (float*)d_out;
    float* spike_out  = out;                                           // (8,20,64,32,32)
    float* count_out  = out + (size_t)N_*T_*COUT_*OH_*OW_;             // (8,64,32,32)
    float* ann_out    = count_out + (size_t)N_*COUT_*OH_*OW_;          // (8,64,32,32)

    prep_weights<<<dim3((COUT_*KDIM + 255) / 256), dim3(256), 0, stream>>>(
        wgt, bias, gamma, beta, rmean, rvar, wS, wA, bS, bA);

    dim3 grid(16, N_);
    conv_pool_kernel<true ><<<grid, dim3(NTHREADS), SMEM_BYTES, stream>>>(
        x_st, wS, bS, spike_out, count_out);
    conv_pool_kernel<false><<<grid, dim3(NTHREADS), SMEM_BYTES, stream>>>(
        x_sc, wA, bA, ann_out, nullptr);
}
